// CausalAttention_78134045049468
// MI455X (gfx1250) — matrix operand order repaired and checked by reading, not compiled, not run on hardware
//
#include <hip/hip_runtime.h>
#include <hip/hip_bf16.h>
#include <math.h>

// ---------------------------------------------------------------------------
// Causal attention for MI455X (gfx1250), bf16 WMMA pipeline:
//   1) Q/K/V = x @ W{q,k,v}   (f32 -> bf16 convert, f32 accumulate via WMMA)
//   2) S = mask(Q K^T / 32)   (f32, per batch, fits in L2)
//   3) P = softmax(S) (bf16)
//   4) O = P @ V              (f32 out, k-loop cut at causal diagonal)
// All GEMMs: 128x128x32 block tile, 4 waves, 64x64 per wave (16 WMMAs/step),
// double-buffered LDS staging with vectorized (b128) global loads.
// Workspace: Q,K,V bf16 (48MB) + S f32 (16MB) + P bf16 (8MB) = 72MB.
// ---------------------------------------------------------------------------

typedef __attribute__((ext_vector_type(16))) __bf16 v16bf;
typedef __attribute__((ext_vector_type(8)))  float  v8f;

#define TILE_M 128
#define TILE_N 128
#define TILE_K 32
#define NTHR   128

union F4 { float4 v; float f[4]; };
union U4 { uint4 v; unsigned int u[4]; unsigned short us[8]; };
union BF2 { __bf16 h[2]; unsigned int u; };
union BF4 { __bf16 h[4]; uint2 u2; };

// ------------------------- WMMA fragment loaders ---------------------------
// A-fragment (16x32 bf16), ISA 7.12.2: lane<16 -> M=lane, K {0..7,16..23};
// lane>=16 -> same row, K {8..15,24..31}. Two contiguous 16B runs per lane.
__device__ __forceinline__ v16bf frag_a(const unsigned int* lds, int row0, int lane) {
  const int m  = lane & 15;
  const int kh = (lane >> 4) & 1;
  union { v16bf v; unsigned int u[8]; } f;
#pragma unroll
  for (int r = 0; r < 8; ++r) {
    const int k = 2 * (r & 3) + 16 * (r >> 2) + 8 * kh;
    f.u[r] = lds[(row0 + m) * (TILE_K / 2) + (k >> 1)];
  }
  return f.v;
}

// B-fragment (32x16 bf16): lane<16 -> col N=lane, K {0..7,16..23}; lane>=16 ->
// same col, K {8..15,24..31}. LDS holds B^T tile as [n][k] -> two 16B runs.
__device__ __forceinline__ v16bf frag_b(const unsigned int* lds, int col0, int lane) {
  const int n  = lane & 15;
  const int kh = (lane >> 4) & 1;
  union { v16bf v; unsigned int u[8]; } f;
#pragma unroll
  for (int r = 0; r < 8; ++r) {
    const int k = 2 * (r & 3) + 16 * (r >> 2) + 8 * kh;
    f.u[r] = lds[(col0 + n) * (TILE_K / 2) + (k >> 1)];
  }
  return f.v;
}

__device__ __forceinline__ void zero_acc(v8f acc[4][4]) {
#pragma unroll
  for (int i = 0; i < 4; ++i)
#pragma unroll
    for (int j = 0; j < 4; ++j)
#pragma unroll
      for (int e = 0; e < 8; ++e) acc[i][j][e] = 0.0f;
}

__device__ __forceinline__ void wave_mma(const __bf16* sa, const __bf16* sb,
                                         int wm, int wn, int lane, v8f acc[4][4]) {
  const unsigned int* saU = (const unsigned int*)sa;
  const unsigned int* sbU = (const unsigned int*)sb;
  v16bf af[4], bfrag[4];
#pragma unroll
  for (int i = 0; i < 4; ++i) af[i] = frag_a(saU, wm + i * 16, lane);
#pragma unroll
  for (int j = 0; j < 4; ++j) bfrag[j] = frag_b(sbU, wn + j * 16, lane);
#pragma unroll
  for (int i = 0; i < 4; ++i)
#pragma unroll
    for (int j = 0; j < 4; ++j)
      acc[i][j] = __builtin_amdgcn_wmma_f32_16x16x32_bf16(
          false, af[i], false, bfrag[j], (short)0, acc[i][j], false, false);
}

// ------------------------- vectorized tile staging -------------------------
// f32 [row][k] tile (128x32) -> bf16 LDS [row][k]; 8x float4 per thread.
struct RegsAf32 { F4 v[8]; };
__device__ __forceinline__ RegsAf32 fetch_a_f32(const float* A, int ld, int m0, int k0, int tid) {
  RegsAf32 r;
#pragma unroll
  for (int i = 0; i < 8; ++i) {
    const int slot = tid + i * NTHR;            // 0..1023
    const int row = slot >> 3, c4 = slot & 7;
    r.v[i].v = *(const float4*)(A + (size_t)(m0 + row) * ld + k0 + c4 * 4);
  }
  return r;
}
__device__ __forceinline__ void store_a_f32(const RegsAf32& r, __bf16* sa, int tid) {
#pragma unroll
  for (int i = 0; i < 8; ++i) {
    const int slot = tid + i * NTHR;
    const int row = slot >> 3, c4 = slot & 7;
    BF4 p;
#pragma unroll
    for (int j = 0; j < 4; ++j) p.h[j] = (__bf16)r.v[i].f[j];
    *(uint2*)(sa + row * TILE_K + c4 * 4) = p.u2;
  }
}

// f32 [k][n] tile (32x128) -> bf16 LDS [n][k] (transpose); row-pair packing.
struct RegsWf32 { F4 a[4], b[4]; };
__device__ __forceinline__ RegsWf32 fetch_w_f32(const float* W, int ld, int k0, int n0, int tid) {
  RegsWf32 r;
#pragma unroll
  for (int i = 0; i < 4; ++i) {
    const int slot = tid + i * NTHR;            // 0..511
    const int kp = slot >> 5, c4 = slot & 31;
    r.a[i].v = *(const float4*)(W + (size_t)(k0 + 2 * kp)     * ld + n0 + c4 * 4);
    r.b[i].v = *(const float4*)(W + (size_t)(k0 + 2 * kp + 1) * ld + n0 + c4 * 4);
  }
  return r;
}
__device__ __forceinline__ void store_w_f32(const RegsWf32& r, __bf16* sb, int tid) {
#pragma unroll
  for (int i = 0; i < 4; ++i) {
    const int slot = tid + i * NTHR;
    const int kp = slot >> 5, c4 = slot & 31;
#pragma unroll
    for (int j = 0; j < 4; ++j) {
      BF2 p;
      p.h[0] = (__bf16)r.a[i].f[j];
      p.h[1] = (__bf16)r.b[i].f[j];
      *(unsigned int*)(sb + (c4 * 4 + j) * TILE_K + 2 * kp) = p.u;
    }
  }
}

// bf16 row-major [row][k] tile (128x32) -> LDS [row][k]; 4x uint4 per thread.
struct RegsB16 { uint4 v[4]; };
__device__ __forceinline__ RegsB16 fetch_bf16(const __bf16* G, int ld, int r0, int k0, int tid) {
  RegsB16 r;
#pragma unroll
  for (int i = 0; i < 4; ++i) {
    const int slot = tid + i * NTHR;            // 0..511
    const int row = slot >> 2, c = slot & 3;
    r.v[i] = *(const uint4*)(G + (size_t)(r0 + row) * ld + k0 + c * 8);
  }
  return r;
}
__device__ __forceinline__ void store_bf16(const RegsB16& r, __bf16* s, int tid) {
#pragma unroll
  for (int i = 0; i < 4; ++i) {
    const int slot = tid + i * NTHR;
    const int row = slot >> 2, c = slot & 3;
    *(uint4*)(s + row * TILE_K + c * 8) = r.v[i];
  }
}

// bf16 [k][n] tile (32x128) -> LDS [n][k] (transpose); row-pair packing.
struct RegsVT { U4 a[2], b[2]; };
__device__ __forceinline__ RegsVT fetch_vt(const __bf16* V, int ld, int k0, int n0, int tid) {
  RegsVT r;
#pragma unroll
  for (int i = 0; i < 2; ++i) {
    const int slot = tid + i * NTHR;            // 0..255
    const int kp = slot >> 4, g = slot & 15;
    r.a[i].v = *(const uint4*)(V + (size_t)(k0 + 2 * kp)     * ld + n0 + g * 8);
    r.b[i].v = *(const uint4*)(V + (size_t)(k0 + 2 * kp + 1) * ld + n0 + g * 8);
  }
  return r;
}
__device__ __forceinline__ void store_vt(const RegsVT& r, __bf16* sb, int tid) {
#pragma unroll
  for (int i = 0; i < 2; ++i) {
    const int slot = tid + i * NTHR;
    const int kp = slot >> 4, g = slot & 15;
#pragma unroll
    for (int j = 0; j < 8; ++j) {
      const unsigned int u = (unsigned int)r.a[i].us[j] |
                             ((unsigned int)r.b[i].us[j] << 16);
      *(unsigned int*)(sb + (g * 8 + j) * TILE_K + 2 * kp) = u;
    }
  }
}

// -------------------- 1) projections: C_bf16 = A_f32 @ W_f32 ---------------
__global__ __launch_bounds__(128) void proj_kernel(const float* __restrict__ A,
                                                   const float* __restrict__ W,
                                                   __bf16* __restrict__ C,
                                                   int M, int N, int K) {
  __shared__ __align__(16) __bf16 sa[2][TILE_M * TILE_K];
  __shared__ __align__(16) __bf16 sb[2][TILE_N * TILE_K];
  const int tid  = threadIdx.x;
  const int lane = tid & 31;
  const int wave = tid >> 5;
  const int wm = (wave >> 1) * 64;
  const int wn = (wave & 1) * 64;
  const int m0 = blockIdx.y * TILE_M;
  const int n0 = blockIdx.x * TILE_N;

  v8f acc[4][4];
  zero_acc(acc);

  RegsAf32 ra = fetch_a_f32(A, K, m0, 0, tid);
  RegsWf32 rw = fetch_w_f32(W, N, 0, n0, tid);
  store_a_f32(ra, sa[0], tid);
  store_w_f32(rw, sb[0], tid);
  __syncthreads();

  int buf = 0;
  for (int k0 = 0; k0 < K; k0 += TILE_K) {
    const bool more = (k0 + TILE_K) < K;
    if (more) {
      ra = fetch_a_f32(A, K, m0, k0 + TILE_K, tid);
      rw = fetch_w_f32(W, N, k0 + TILE_K, n0, tid);
    }
    wave_mma(sa[buf], sb[buf], wm, wn, lane, acc);
    if (more) {
      store_a_f32(ra, sa[buf ^ 1], tid);
      store_w_f32(rw, sb[buf ^ 1], tid);
    }
    __syncthreads();
    buf ^= 1;
  }

  const int n  = lane & 15;
  const int mb = (lane >> 4) * 8;
#pragma unroll
  for (int i = 0; i < 4; ++i)
#pragma unroll
    for (int j = 0; j < 4; ++j)
#pragma unroll
      for (int r = 0; r < 8; ++r) {
        const int row = m0 + wm + i * 16 + mb + r;
        const int col = n0 + wn + j * 16 + n;
        C[row * N + col] = (__bf16)acc[i][j][r];
      }
}

// -------------------- 2) scores: S = mask(Q K^T * scale) -------------------
__global__ __launch_bounds__(128) void scores_kernel(const __bf16* __restrict__ Q,
                                                     const __bf16* __restrict__ Km,
                                                     float* __restrict__ S,
                                                     int Sq, int D, float scale) {
  const int m0  = blockIdx.y * TILE_M;
  const int n0  = blockIdx.x * TILE_N;
  const int tid = threadIdx.x;

  if (n0 >= m0 + TILE_M) {                 // tile entirely above the diagonal
    const float4 fill = make_float4(-1e30f, -1e30f, -1e30f, -1e30f);
    for (int idx = tid; idx < TILE_M * TILE_N / 4; idx += NTHR) {
      const int r = idx >> 5, c4 = idx & 31;
      *(float4*)(S + (size_t)(m0 + r) * Sq + n0 + c4 * 4) = fill;
    }
    return;
  }

  __shared__ __align__(16) __bf16 sa[2][TILE_M * TILE_K];
  __shared__ __align__(16) __bf16 sb[2][TILE_N * TILE_K];
  const int lane = tid & 31;
  const int wave = tid >> 5;
  const int wm = (wave >> 1) * 64;
  const int wn = (wave & 1) * 64;

  v8f acc[4][4];
  zero_acc(acc);

  // K rows are [j][d] == B^T tile layout already: plain contiguous staging.
  RegsB16 rq = fetch_bf16(Q,  D, m0, 0, tid);
  RegsB16 rk = fetch_bf16(Km, D, n0, 0, tid);
  store_bf16(rq, sa[0], tid);
  store_bf16(rk, sb[0], tid);
  __syncthreads();

  int buf = 0;
  for (int k0 = 0; k0 < D; k0 += TILE_K) {
    const bool more = (k0 + TILE_K) < D;
    if (more) {
      rq = fetch_bf16(Q,  D, m0, k0 + TILE_K, tid);
      rk = fetch_bf16(Km, D, n0, k0 + TILE_K, tid);
    }
    wave_mma(sa[buf], sb[buf], wm, wn, lane, acc);
    if (more) {
      store_bf16(rq, sa[buf ^ 1], tid);
      store_bf16(rk, sb[buf ^ 1], tid);
    }
    __syncthreads();
    buf ^= 1;
  }

  const int n  = lane & 15;
  const int mb = (lane >> 4) * 8;
#pragma unroll
  for (int i = 0; i < 4; ++i)
#pragma unroll
    for (int j = 0; j < 4; ++j)
#pragma unroll
      for (int r = 0; r < 8; ++r) {
        const int gi = m0 + wm + i * 16 + mb + r;
        const int gj = n0 + wn + j * 16 + n;
        float v = acc[i][j][r] * scale;
        if (gj > gi) v = -1e30f;
        S[gi * Sq + gj] = v;
      }
}

// -------------------- 3) rowwise softmax, f32 -> bf16 ----------------------
__global__ __launch_bounds__(256) void softmax_kernel(const float* __restrict__ S,
                                                      __bf16* __restrict__ P,
                                                      int Sq) {
  __shared__ float red[256];
  const int row = blockIdx.x;
  const int tid = threadIdx.x;
  const float4* s4 = (const float4*)(S + (size_t)row * Sq);
  const int n4 = Sq >> 2;

  float m = -3.0e38f;
  for (int j = tid; j < n4; j += 256) {
    const float4 v = s4[j];
    m = fmaxf(m, fmaxf(fmaxf(v.x, v.y), fmaxf(v.z, v.w)));
  }
  red[tid] = m;
  __syncthreads();
  for (int w = 128; w > 0; w >>= 1) {
    if (tid < w) red[tid] = fmaxf(red[tid], red[tid + w]);
    __syncthreads();
  }
  const float gmax = red[0];
  __syncthreads();

  float sum = 0.0f;
  for (int j = tid; j < n4; j += 256) {
    const float4 v = s4[j];
    sum += expf(v.x - gmax) + expf(v.y - gmax) + expf(v.z - gmax) + expf(v.w - gmax);
  }
  red[tid] = sum;
  __syncthreads();
  for (int w = 128; w > 0; w >>= 1) {
    if (tid < w) red[tid] += red[tid + w];
    __syncthreads();
  }
  const float inv = 1.0f / red[0];

  __bf16* p = P + (size_t)row * Sq;
  for (int j = tid; j < n4; j += 256) {
    const float4 v = s4[j];
    BF4 o;
    o.h[0] = (__bf16)(expf(v.x - gmax) * inv);
    o.h[1] = (__bf16)(expf(v.y - gmax) * inv);
    o.h[2] = (__bf16)(expf(v.z - gmax) * inv);
    o.h[3] = (__bf16)(expf(v.w - gmax) * inv);
    *(uint2*)(p + j * 4) = o.u2;
  }
}

// -------------------- 4) O_f32 = P_bf16 @ V_bf16 (causal k-cut) ------------
__global__ __launch_bounds__(128) void pv_kernel(const __bf16* __restrict__ P,
                                                 const __bf16* __restrict__ V,
                                                 float* __restrict__ O,
                                                 int Sq, int D) {
  __shared__ __align__(16) __bf16 sa[2][TILE_M * TILE_K];
  __shared__ __align__(16) __bf16 sb[2][TILE_N * TILE_K];
  const int tid  = threadIdx.x;
  const int lane = tid & 31;
  const int wave = tid >> 5;
  const int wm = (wave >> 1) * 64;
  const int wn = (wave & 1) * 64;
  const int m0 = blockIdx.y * TILE_M;
  const int n0 = blockIdx.x * TILE_N;

  v8f acc[4][4];
  zero_acc(acc);

  const int kend = (Sq < m0 + TILE_M) ? Sq : (m0 + TILE_M);  // P is 0 past diag
  RegsB16 rp = fetch_bf16(P, Sq, m0, 0, tid);
  RegsVT  rv = fetch_vt(V, D, 0, n0, tid);
  store_bf16(rp, sa[0], tid);
  store_vt(rv, sb[0], tid);
  __syncthreads();

  int buf = 0;
  for (int k0 = 0; k0 < kend; k0 += TILE_K) {
    const bool more = (k0 + TILE_K) < kend;
    if (more) {
      rp = fetch_bf16(P, Sq, m0, k0 + TILE_K, tid);
      rv = fetch_vt(V, D, k0 + TILE_K, n0, tid);
    }
    wave_mma(sa[buf], sb[buf], wm, wn, lane, acc);
    if (more) {
      store_bf16(rp, sa[buf ^ 1], tid);
      store_vt(rv, sb[buf ^ 1], tid);
    }
    __syncthreads();
    buf ^= 1;
  }

  const int n  = lane & 15;
  const int mb = (lane >> 4) * 8;
#pragma unroll
  for (int i = 0; i < 4; ++i)
#pragma unroll
    for (int j = 0; j < 4; ++j)
#pragma unroll
      for (int r = 0; r < 8; ++r) {
        const int row = m0 + wm + i * 16 + mb + r;
        const int col = n0 + wn + j * 16 + n;
        O[row * D + col] = acc[i][j][r];
      }
}

// ---------------------------------------------------------------------------
extern "C" void kernel_launch(void* const* d_in, const int* in_sizes, int n_in,
                              void* d_out, int out_size, void* d_ws, size_t ws_size,
                              hipStream_t stream) {
  (void)in_sizes; (void)n_in; (void)out_size; (void)ws_size;
  const float* x  = (const float*)d_in[0];
  const float* Wq = (const float*)d_in[1];
  const float* Wk = (const float*)d_in[2];
  const float* Wv = (const float*)d_in[3];

  const int B = 4, S = 2048, D = 1024;
  const int Mtot = B * S;                      // 8192 tokens

  char* w = (char*)d_ws;
  __bf16* Qb = (__bf16*)(w);
  __bf16* Kb = (__bf16*)(w + (size_t)Mtot * D * 2);
  __bf16* Vb = (__bf16*)(w + (size_t)Mtot * D * 4);
  float*  Sc = (float*) (w + (size_t)Mtot * D * 6);                      // 16 MB
  __bf16* P  = (__bf16*)(w + (size_t)Mtot * D * 6 + (size_t)S * S * 4);  //  8 MB

  dim3 blk(128);
  dim3 gproj(D / TILE_N, Mtot / TILE_M);       // (8, 64)
  proj_kernel<<<gproj, blk, 0, stream>>>(x, Wq, Qb, Mtot, D, D);
  proj_kernel<<<gproj, blk, 0, stream>>>(x, Wk, Kb, Mtot, D, D);
  proj_kernel<<<gproj, blk, 0, stream>>>(x, Wv, Vb, Mtot, D, D);

  const float scale = 1.0f / 32.0f;            // 1/sqrt(1024)
  dim3 gsc(S / TILE_N, S / TILE_M);            // (16, 16)
  dim3 gpv(D / TILE_N, S / TILE_M);            // (8, 16)
  for (int b = 0; b < B; ++b) {
    const __bf16* Qb_b = Qb + (size_t)b * S * D;
    const __bf16* Kb_b = Kb + (size_t)b * S * D;
    const __bf16* Vb_b = Vb + (size_t)b * S * D;
    float* O_b = (float*)d_out + (size_t)b * S * D;
    scores_kernel<<<gsc, blk, 0, stream>>>(Qb_b, Kb_b, Sc, S, D, scale);
    softmax_kernel<<<dim3(S), dim3(256), 0, stream>>>(Sc, P, S);
    pv_kernel<<<gpv, blk, 0, stream>>>(P, Vb_b, O_b, S, D);
  }
}
